// PeriodicSAModule2_609885356790
// MI455X (gfx1250) — matrix operand order repaired and checked by reading, not compiled, not run on hardware
//
#include <hip/hip_runtime.h>
#include <hip/hip_bf16.h>

typedef __attribute__((ext_vector_type(16))) _Float16 v16h;
typedef __attribute__((ext_vector_type(8)))  _Float16 v8h;
typedef __attribute__((ext_vector_type(8)))  float    v8f;

#define F_IN   64
#define HDIM   128
#define F_OUT  128
#define WAVES  4          // waves per block (128 threads)

// d_ws layout (as _Float16):
//   [0, 12288)        : W1 ready, [kk=3][nt=8][lane=32][j=16]  (B-matrix layout, K padded 67->96)
//   [12288, 28672)    : W2 ready, [kk=4][nt=8][lane=32][j=16]
#define W1_HALFS 12288
#define W2_HALFS 16384

// ---------------- weight prep: f32 -> f16, pre-swizzled to WMMA B layout ----
// B-matrix (32x16 f16, wave32): lane L holds column N = L%16,
// element j (0..15) holds K = 32*kk + 8*(L/16) + (j&7) + 16*(j>>3).
__global__ void prep_weights_kernel(const float* __restrict__ W1,
                                    const float* __restrict__ W2,
                                    _Float16* __restrict__ ws) {
    int t = blockIdx.x * 256 + threadIdx.x;
    if (t < W1_HALFS) {
        int j    = t & 15;
        int lane = (t >> 4) & 31;
        int nt   = (t >> 9) & 7;
        int kk   = t >> 12;                       // 0..2
        int n    = nt * 16 + (lane & 15);
        int K    = kk * 32 + (lane >> 4) * 8 + (j & 7) + (j >> 3) * 16;
        ws[t] = (K < F_IN + 3) ? (_Float16)W1[K * HDIM + n] : (_Float16)0.0f;
    } else if (t < W1_HALFS + W2_HALFS) {
        int u    = t - W1_HALFS;
        int j    = u & 15;
        int lane = (u >> 4) & 31;
        int nt   = (u >> 9) & 7;
        int kk   = u >> 12;                       // 0..3
        int n    = nt * 16 + (lane & 15);
        int K    = kk * 32 + (lane >> 4) * 8 + (j & 7) + (j >> 3) * 16;
        ws[t] = (_Float16)W2[K * F_OUT + n];
    }
}

// ---------------- output init: segment_max identity for relu'd data is 0 ----
__global__ void init_out_kernel(float* __restrict__ out, int n) {
    int i = blockIdx.x * 256 + threadIdx.x;
    if (i < n) out[i] = 0.0f;
}

__device__ __forceinline__ v16h pack16(float4 a, float4 b, float4 c, float4 d) {
    v16h r;
    r[0]=(_Float16)a.x;  r[1]=(_Float16)a.y;  r[2]=(_Float16)a.z;  r[3]=(_Float16)a.w;
    r[4]=(_Float16)b.x;  r[5]=(_Float16)b.y;  r[6]=(_Float16)b.z;  r[7]=(_Float16)b.w;
    r[8]=(_Float16)c.x;  r[9]=(_Float16)c.y;  r[10]=(_Float16)c.z; r[11]=(_Float16)c.w;
    r[12]=(_Float16)d.x; r[13]=(_Float16)d.y; r[14]=(_Float16)d.z; r[15]=(_Float16)d.w;
    return r;
}

// ---------------- main edge kernel: 16 edges per wave, WMMA MLP + atomic max
__global__ void __launch_bounds__(WAVES * 32)
edge_mlp_kernel(const float* __restrict__ x,
                const float* __restrict__ pos,
                const float* __restrict__ trans_vec,
                const float* __restrict__ scale,
                const float* __restrict__ offset,
                const float* __restrict__ b1,
                const float* __restrict__ b2,
                const int*   __restrict__ batch,
                const int*   __restrict__ row,
                const int*   __restrict__ expand,
                const _Float16* __restrict__ wready,
                unsigned int* __restrict__ out,
                int ntiles) {
    __shared__ _Float16 sW1[W1_HALFS];            // 24 KB
    __shared__ _Float16 sW2[W2_HALFS];            // 32 KB
    __shared__ _Float16 sBuf[WAVES][16 * 32];     //  4 KB transpose bounce

    const int tid = threadIdx.x;

    // Stage pre-swizzled weights into LDS with b128 copies.
    {
        const uint4* src = (const uint4*)wready;
        uint4* d1 = (uint4*)sW1;
        uint4* d2 = (uint4*)sW2;
        #pragma unroll 4
        for (int i = tid; i < W1_HALFS / 8; i += WAVES * 32) d1[i] = src[i];
        #pragma unroll 4
        for (int i = tid; i < W2_HALFS / 8; i += WAVES * 32) d2[i] = src[W1_HALFS / 8 + i];
    }
    __syncthreads();

    const int wave = tid >> 5;
    const int lane = tid & 31;
    const int tile = blockIdx.x * WAVES + wave;
    if (tile >= ntiles) return;

    const int m16  = lane & 15;       // this lane's edge-row within the tile
    const int half = lane >> 4;       // A-layout K-group select
    const int kb   = half * 8;
    const int e    = tile * 16 + m16;

    // ---- per-edge geometry: rel = pos[src] + offset*lat[graph] - pos[dst]
    const int ri = row[e];
    const int si = expand[e];
    const int bg = batch[ri];
    const float sc = scale[bg];
    const float* tv = trans_vec + bg * 9;
    float rel[3];
    #pragma unroll
    for (int d = 0; d < 3; ++d) {
        float lat = (tv[3 * d] + tv[3 * d + 1] + tv[3 * d + 2]) * sc;
        rel[d] = pos[3 * si + d] + offset[3 * (long)e + d] * lat - pos[3 * ri + d];
    }

    // ---- layer 1: [16 x 96] @ [96 x 128]  (3 K-steps x 8 N-tiles)
    v8f acc1[8];
    #pragma unroll
    for (int nt = 0; nt < 8; ++nt) {
        float bv = b1[nt * 16 + m16];
        acc1[nt] = (v8f){bv, bv, bv, bv, bv, bv, bv, bv};
    }

    const float* xs = x + (long)si * F_IN;
    #pragma unroll
    for (int kk = 0; kk < 2; ++kk) {
        // A-layout: lane holds K = 32*kk + kb + {0..7} and 32*kk + 16 + kb + {0..7}
        const float4* p = (const float4*)(xs + kk * 32 + kb);
        const float4* q = (const float4*)(xs + kk * 32 + kb + 16);
        float4 a0 = p[0], a1 = p[1], a2 = q[0], a3 = q[1];
        v16h A = pack16(a0, a1, a2, a3);
        #pragma unroll
        for (int nt = 0; nt < 8; ++nt) {
            v16h B = *(const v16h*)&sW1[((kk * 8 + nt) * 32 + lane) * 16];
            acc1[nt] = __builtin_amdgcn_wmma_f32_16x16x32_f16(
                false, A, false, B, (short)0, acc1[nt], false, false);
        }
    }
    {   // K-step 2: K = 64..95, only K=64..66 (rel) nonzero, rest padded zero
        v16h A;
        #pragma unroll
        for (int j = 0; j < 16; ++j) A[j] = (_Float16)0.0f;
        if (half == 0) {
            A[0] = (_Float16)rel[0];
            A[1] = (_Float16)rel[1];
            A[2] = (_Float16)rel[2];
        }
        #pragma unroll
        for (int nt = 0; nt < 8; ++nt) {
            v16h B = *(const v16h*)&sW1[((2 * 8 + nt) * 32 + lane) * 16];
            acc1[nt] = __builtin_amdgcn_wmma_f32_16x16x32_f16(
                false, A, false, B, (short)0, acc1[nt], false, false);
        }
    }
    #pragma unroll
    for (int nt = 0; nt < 8; ++nt)
        #pragma unroll
        for (int r = 0; r < 8; ++r)
            acc1[nt][r] = fmaxf(acc1[nt][r], 0.0f);

    // ---- layer 2: [16 x 128] @ [128 x 128], h1 re-laid out via LDS bounce
    v8f acc2[8];
    #pragma unroll
    for (int nt = 0; nt < 8; ++nt) {
        float bv = b2[nt * 16 + m16];
        acc2[nt] = (v8f){bv, bv, bv, bv, bv, bv, bv, bv};
    }

    _Float16* buf = &sBuf[wave][0];   // [M=16][K=32] row-major chunk
    #pragma unroll
    for (int kk = 0; kk < 4; ++kk) {
        // scatter C-layout (M = r + 8*half, N = lane&15) into row-major f16
        #pragma unroll
        for (int t = 0; t < 2; ++t) {
            v8f hv = acc1[2 * kk + t];
            #pragma unroll
            for (int r = 0; r < 8; ++r)
                buf[(r + half * 8) * 32 + t * 16 + m16] = (_Float16)hv[r];
        }
        // per-wave LDS is in-order, but fence the compiler too
        asm volatile("s_wait_dscnt 0" ::: "memory");
        // gather back in A layout: two contiguous 8-half runs per lane
        v8h lo = *(const v8h*)&buf[m16 * 32 + kb];
        v8h hi = *(const v8h*)&buf[m16 * 32 + 16 + kb];
        v16h A = __builtin_shufflevector(lo, hi,
                                         0, 1, 2, 3, 4, 5, 6, 7,
                                         8, 9, 10, 11, 12, 13, 14, 15);
        #pragma unroll
        for (int nt = 0; nt < 8; ++nt) {
            v16h B = *(const v16h*)&sW2[((kk * 8 + nt) * 32 + lane) * 16];
            acc2[nt] = __builtin_amdgcn_wmma_f32_16x16x32_f16(
                false, A, false, B, (short)0, acc2[nt], false, false);
        }
    }

    // ---- relu + segment max via u32 atomicMax (all values >= 0 after relu)
    int rws[8];
    #pragma unroll
    for (int r = 0; r < 8; ++r) rws[r] = row[tile * 16 + r + half * 8];

    #pragma unroll
    for (int nt = 0; nt < 8; ++nt) {
        #pragma unroll
        for (int r = 0; r < 8; ++r) {
            float v = fmaxf(acc2[nt][r], 0.0f);
            if (v > 0.0f) {
                atomicMax(out + (long)rws[r] * F_OUT + nt * 16 + m16,
                          __float_as_uint(v));
            }
        }
    }
}

extern "C" void kernel_launch(void* const* d_in, const int* in_sizes, int n_in,
                              void* d_out, int out_size, void* d_ws, size_t ws_size,
                              hipStream_t stream) {
    const float* x         = (const float*)d_in[0];
    const float* pos       = (const float*)d_in[1];
    // d_in[2] fps_pos, d_in[3] frac_pos: unused by the reference math
    const float* trans_vec = (const float*)d_in[4];
    const float* scale     = (const float*)d_in[5];
    const float* offset    = (const float*)d_in[6];
    const float* W1        = (const float*)d_in[7];
    const float* b1        = (const float*)d_in[8];
    const float* W2        = (const float*)d_in[9];
    const float* b2        = (const float*)d_in[10];
    const int*   batch     = (const int*)d_in[11];
    const int*   row       = (const int*)d_in[12];
    const int*   expand    = (const int*)d_in[13];

    const int E = in_sizes[12];
    const int ntiles = E / 16;

    _Float16* wready = (_Float16*)d_ws;

    // 1) zero output (segment-max identity; empty segments -> 0)
    {
        int n = out_size;
        init_out_kernel<<<(n + 255) / 256, 256, 0, stream>>>((float*)d_out, n);
    }
    // 2) swizzle weights into WMMA-B-ready f16 layout in workspace
    {
        int total = W1_HALFS + W2_HALFS;
        prep_weights_kernel<<<(total + 255) / 256, 256, 0, stream>>>(W1, W2, wready);
    }
    // 3) edge-tiled WMMA MLP + atomic segment max
    {
        int blocks = (ntiles + WAVES - 1) / WAVES;
        edge_mlp_kernel<<<blocks, WAVES * 32, 0, stream>>>(
            x, pos, trans_vec, scale, offset, b1, b2,
            batch, row, expand, wready, (unsigned int*)d_out, ntiles);
    }
}
